// MLAAttentionJoint_24644522344789
// MI455X (gfx1250) — matrix operand order repaired and checked by reading, not compiled, not run on hardware
//
#include <hip/hip_runtime.h>
#include <hip/hip_bf16.h>

// ---------------- problem constants ----------------
#define DIM   2048   // hidden
#define NH    16     // heads
#define DHD   128    // head dim
#define RQD   1024   // q rank
#define RKD   512    // kv rank
#define BBATCH 2
#define TSEQ  2048

typedef unsigned short u16;
typedef __attribute__((ext_vector_type(16))) __bf16 v16bf;
typedef __attribute__((ext_vector_type(8)))  float  v8f;

union ABFrag { v16bf v; uint4 q[2]; u16 s[16]; };

__device__ __forceinline__ u16 f2bf(float f) {
  unsigned u = __float_as_uint(f);
  u += 0x7FFFu + ((u >> 16) & 1u);      // round-to-nearest-even
  return (u16)(u >> 16);
}

// A fragment (16x32 bf16): lane owns row; j<8 -> K=kk+j+8*hi ; j>=8 -> K=kk+16+(j-8)+8*hi
__device__ __forceinline__ void ld_afrag(const u16* __restrict__ row, int kk, int hi, ABFrag& f) {
  f.q[0] = *(const uint4*)(row + kk + hi * 8);
  f.q[1] = *(const uint4*)(row + kk + 16 + hi * 8);
}
// B fragment (32x16 bf16 of B^T row): lane owns col; j<8 -> K=kk+j+8*hi ; j>=8 -> K=kk+16+(j-8)+8*hi
__device__ __forceinline__ void ld_bfrag(const u16* __restrict__ row, int kk, int hi, ABFrag& f) {
  f.q[0] = *(const uint4*)(row + kk + hi * 8);
  f.q[1] = *(const uint4*)(row + kk + 16 + hi * 8);
}

// ---- CDNA5 async global->LDS (ASYNCcnt-tracked) ----
typedef __attribute__((address_space(3))) const void lds_cv;
__device__ __forceinline__ unsigned lds_addr(const void* p) {
  return (unsigned)(size_t)(lds_cv*)p;           // generic -> LDS byte offset
}
__device__ __forceinline__ void async_copy_b128(const u16* g, unsigned loff) {
  asm volatile("global_load_async_to_lds_b128 %0, %1, off"
               :: "v"(loff), "v"(g) : "memory");
}
__device__ __forceinline__ void wait_async0() {
  asm volatile("s_wait_asynccnt 0x0" ::: "memory");
}

// ---------------- elementwise fp32 -> bf16 ----------------
__global__ void k_f32_to_bf16(const float* __restrict__ in, u16* __restrict__ out, long n) {
  long i = (long)blockIdx.x * blockDim.x + threadIdx.x;
  long st = (long)gridDim.x * blockDim.x;
  for (; i < n; i += st) out[i] = f2bf(in[i]);
}

// out[h][r][d] = W[(h*DHD + d)*R + r]   (per-head transpose to make K contiguous)
__global__ void k_head_transpose(const float* __restrict__ W, u16* __restrict__ out, int R, long n) {
  long i = (long)blockIdx.x * blockDim.x + threadIdx.x;
  long st = (long)gridDim.x * blockDim.x;
  for (; i < n; i += st) {
    int d = (int)(i % DHD);
    long rem = i / DHD;
    int r = (int)(rem % R);
    int h = (int)(rem / R);
    out[i] = f2bf(W[((long)(h * DHD + d)) * R + r]);
  }
}

// ---------------- batched NT GEMM: C[z] = scale * A[z] * B[z]^T (+bias) ----------------
// Block = 256 threads = 8 waves; block tile 128(M) x 64(N); wave tile 16 x 64.
// K-loop software-pipelined (double-buffered fragments, unroll-by-2). Requires K % 64 == 0.
__global__ void k_gemm_nt(const u16* __restrict__ A, const u16* __restrict__ B,
                          void* __restrict__ C,
                          int K, int lda, int ldb, int ldc,
                          long sA, int dA, int mA,
                          long sB, int dB, int mB,
                          long sC, float scale, const float* __restrict__ bias, int c_f32)
{
  const int lane = threadIdx.x & 31;
  const int w    = threadIdx.x >> 5;
  const int hi   = lane >> 4;
  const int ln   = lane & 15;
  const int m0   = blockIdx.y * 128 + w * 16;
  const int n0   = blockIdx.x * 64;
  const int z    = blockIdx.z;

  const u16* a = A + (long)((z / dA) % mA) * sA;
  const u16* b = B + (long)((z / dB) % mB) * sB;
  const u16* arow = a + (size_t)(m0 + ln) * lda;
  const u16* brows[4];
  #pragma unroll
  for (int t = 0; t < 4; t++) brows[t] = b + (size_t)(n0 + t * 16 + ln) * ldb;

  v8f acc[4];
  #pragma unroll
  for (int t = 0; t < 4; t++)
    #pragma unroll
    for (int e = 0; e < 8; e++) acc[t][e] = 0.f;

  ABFrag a0, a1, b0[4], b1[4];
  ld_afrag(arow, 0, hi, a0);
  #pragma unroll
  for (int t = 0; t < 4; t++) ld_bfrag(brows[t], 0, hi, b0[t]);

  int kk = 0;
  for (; kk + 64 < K; kk += 64) {
    ld_afrag(arow, kk + 32, hi, a1);
    #pragma unroll
    for (int t = 0; t < 4; t++) ld_bfrag(brows[t], kk + 32, hi, b1[t]);
    __builtin_prefetch(arow + kk + 128, 0, 1);
    __builtin_prefetch(brows[0] + kk + 128, 0, 1);
    #pragma unroll
    for (int t = 0; t < 4; t++)
      acc[t] = __builtin_amdgcn_wmma_f32_16x16x32_bf16(false, a0.v, false, b0[t].v,
                                                       (short)0, acc[t], false, false);
    ld_afrag(arow, kk + 64, hi, a0);
    #pragma unroll
    for (int t = 0; t < 4; t++) ld_bfrag(brows[t], kk + 64, hi, b0[t]);
    #pragma unroll
    for (int t = 0; t < 4; t++)
      acc[t] = __builtin_amdgcn_wmma_f32_16x16x32_bf16(false, a1.v, false, b1[t].v,
                                                       (short)0, acc[t], false, false);
  }
  ld_afrag(arow, kk + 32, hi, a1);
  #pragma unroll
  for (int t = 0; t < 4; t++) ld_bfrag(brows[t], kk + 32, hi, b1[t]);
  #pragma unroll
  for (int t = 0; t < 4; t++)
    acc[t] = __builtin_amdgcn_wmma_f32_16x16x32_bf16(false, a0.v, false, b0[t].v,
                                                     (short)0, acc[t], false, false);
  #pragma unroll
  for (int t = 0; t < 4; t++)
    acc[t] = __builtin_amdgcn_wmma_f32_16x16x32_bf16(false, a1.v, false, b1[t].v,
                                                     (short)0, acc[t], false, false);

  float bv[4];
  #pragma unroll
  for (int t = 0; t < 4; t++) bv[t] = bias ? bias[n0 + t * 16 + ln] : 0.f;

  const long zc = (long)z * sC;
  #pragma unroll
  for (int t = 0; t < 4; t++) {
    #pragma unroll
    for (int r = 0; r < 8; r++) {
      long idx = zc + (long)(m0 + r + 8 * hi) * ldc + (n0 + t * 16 + ln);
      float v = acc[t][r] * scale + bv[t];
      if (c_f32) ((float*)C)[idx] = v;
      else       ((u16*)C)[idx] = f2bf(v);
    }
  }
}

// ---------------- flash attention over latent space ----------------
// 256 threads = 8 waves per block; block owns 128 query rows of one (b,h).
// Each 32-key chunk of Ckv (32x512) and Vup^T (128x32) is async-staged into LDS
// ONCE per block (8x reuse vs per-wave streaming), double-buffered on ASYNCcnt.
#define KBUF_SZ  (32 * 512)                     // u16 per K buffer  (32 KB)
#define VBUF_SZ  (128 * 32)                     // u16 per V buffer  ( 8 KB)
#define PBUF_OFF (2 * KBUF_SZ + 2 * VBUF_SZ)    // u16 offset of P staging
#define FLASH_LDS_BYTES ((PBUF_OFF + 8 * 16 * 32) * 2)   // 90112 B = 88 KB

__device__ __forceinline__ void flash_stage(int tid, const u16* __restrict__ kvb,
                                            const u16* __restrict__ vtb, int s0,
                                            u16* kb, u16* vb) {
  #pragma unroll
  for (int j = 0; j < 8; j++) {                 // 32x512 u16 = 2048 x b128
    int idx = tid * 8 + j;
    int row = idx >> 6, seg = idx & 63;
    async_copy_b128(kvb + (size_t)(s0 + row) * RKD + seg * 8,
                    lds_addr(kb + row * 512 + seg * 8));
  }
  #pragma unroll
  for (int j = 0; j < 2; j++) {                 // 128x32 u16 = 512 x b128
    int idx = tid * 2 + j;
    int row = idx >> 2, seg = idx & 3;
    async_copy_b128(vtb + (size_t)row * TSEQ + s0 + seg * 8,
                    lds_addr(vb + row * 32 + seg * 8));
  }
}

__global__ void __launch_bounds__(256)
k_flash(const u16* __restrict__ Qm, const u16* __restrict__ Ckv,
        const u16* __restrict__ VupT, u16* __restrict__ ctxb)
{
  extern __shared__ __align__(16) u16 smem[];
  const int tid  = threadIdx.x;
  const int lane = tid & 31;
  const int w    = tid >> 5;
  const int hi = lane >> 4, ln = lane & 15;
  const int bh = blockIdx.y;
  const int b = bh >> 4;        // / NH
  const int h = bh & 15;        // % NH
  const int t0b = blockIdx.x * 128;
  const int t0  = t0b + w * 16;               // this wave's query rows

  const u16* qrow = Qm  + (size_t)bh * TSEQ * RKD + (size_t)(t0 + ln) * RKD;
  const u16* kvb  = Ckv + (size_t)b  * TSEQ * RKD;
  const u16* vtb  = VupT + (size_t)bh * DHD * TSEQ;
  u16* pb = smem + PBUF_OFF + w * (16 * 32);  // per-wave P staging

  v8f ctx[8];
  float mrow[8], lrow[8];
  #pragma unroll
  for (int t = 0; t < 8; t++) {
    mrow[t] = -__builtin_inff(); lrow[t] = 0.f;
    #pragma unroll
    for (int e = 0; e < 8; e++) ctx[t][e] = 0.f;
  }

  const int nch = (t0b + 128) / 32;           // chunks needed by this block
  flash_stage(tid, kvb, vtb, 0, smem, smem + 2 * KBUF_SZ);
  wait_async0();
  __syncthreads();

  for (int c = 0; c < nch; c++) {
    if (c + 1 < nch)
      flash_stage(tid, kvb, vtb, (c + 1) * 32,
                  smem + ((c + 1) & 1) * KBUF_SZ,
                  smem + 2 * KBUF_SZ + ((c + 1) & 1) * VBUF_SZ);

    const int s0 = c * 32;
    const u16* kb = smem + (c & 1) * KBUF_SZ;
    const u16* vb = smem + 2 * KBUF_SZ + (c & 1) * VBUF_SZ;

    if (s0 <= t0 + 15) {                      // wave-uniform causal skip
      v8f c0, c1;
      #pragma unroll
      for (int e = 0; e < 8; e++) { c0[e] = 0.f; c1[e] = 0.f; }

      const u16* kr0 = kb + (size_t)ln * 512;
      const u16* kr1 = kb + (size_t)(16 + ln) * 512;
      ABFrag qA, qB, k0, k1;
      ld_afrag(qrow, 0, hi, qA);
      int kk = 0;
      for (; kk + 64 < RKD; kk += 64) {
        ld_afrag(qrow, kk + 32, hi, qB);
        ld_bfrag(kr0, kk, hi, k0); ld_bfrag(kr1, kk, hi, k1);
        c0 = __builtin_amdgcn_wmma_f32_16x16x32_bf16(false, qA.v, false, k0.v, (short)0, c0, false, false);
        c1 = __builtin_amdgcn_wmma_f32_16x16x32_bf16(false, qA.v, false, k1.v, (short)0, c1, false, false);
        ld_afrag(qrow, kk + 64, hi, qA);
        ld_bfrag(kr0, kk + 32, hi, k0); ld_bfrag(kr1, kk + 32, hi, k1);
        c0 = __builtin_amdgcn_wmma_f32_16x16x32_bf16(false, qB.v, false, k0.v, (short)0, c0, false, false);
        c1 = __builtin_amdgcn_wmma_f32_16x16x32_bf16(false, qB.v, false, k1.v, (short)0, c1, false, false);
      }
      ld_afrag(qrow, kk + 32, hi, qB);
      ld_bfrag(kr0, kk, hi, k0); ld_bfrag(kr1, kk, hi, k1);
      c0 = __builtin_amdgcn_wmma_f32_16x16x32_bf16(false, qA.v, false, k0.v, (short)0, c0, false, false);
      c1 = __builtin_amdgcn_wmma_f32_16x16x32_bf16(false, qA.v, false, k1.v, (short)0, c1, false, false);
      ld_bfrag(kr0, kk + 32, hi, k0); ld_bfrag(kr1, kk + 32, hi, k1);
      c0 = __builtin_amdgcn_wmma_f32_16x16x32_bf16(false, qB.v, false, k0.v, (short)0, c0, false, false);
      c1 = __builtin_amdgcn_wmma_f32_16x16x32_bf16(false, qB.v, false, k1.v, (short)0, c1, false, false);

      float fac[8];
      #pragma unroll
      for (int r = 0; r < 8; r++) {
        int trow = t0 + r + 8 * hi;
        float x0 = (s0 + ln      <= trow) ? c0[r] : -__builtin_inff();
        float x1 = (s0 + 16 + ln <= trow) ? c1[r] : -__builtin_inff();
        float mx = fmaxf(x0, x1);
        #pragma unroll
        for (int o = 1; o < 16; o <<= 1) mx = fmaxf(mx, __shfl_xor(mx, o, 32));
        float nm = fmaxf(mrow[r], mx);
        fac[r] = __expf(mrow[r] - nm);
        float p0 = __expf(x0 - nm), p1 = __expf(x1 - nm);
        float ps = p0 + p1;
        #pragma unroll
        for (int o = 1; o < 16; o <<= 1) ps += __shfl_xor(ps, o, 32);
        lrow[r] = lrow[r] * fac[r] + ps;
        mrow[r] = nm;
        c0[r] = p0; c1[r] = p1;
      }

      #pragma unroll
      for (int t = 0; t < 8; t++)
        #pragma unroll
        for (int r = 0; r < 8; r++) ctx[t][r] *= fac[r];

      // per-wave P transpose through LDS (same-wave DS ops are in-order;
      // asm fence stops compiler reordering across the type-pun)
      #pragma unroll
      for (int r = 0; r < 8; r++) {
        pb[(r + 8 * hi) * 32 + ln]      = f2bf(c0[r]);
        pb[(r + 8 * hi) * 32 + 16 + ln] = f2bf(c1[r]);
      }
      asm volatile("" ::: "memory");
      ABFrag pa;
      pa.q[0] = *(const uint4*)(pb + ln * 32 + 8 * hi);
      pa.q[1] = *(const uint4*)(pb + ln * 32 + 16 + 8 * hi);

      #pragma unroll
      for (int t = 0; t < 8; t++) {
        ABFrag vf;
        ld_bfrag(vb + (size_t)(t * 16 + ln) * 32, 0, hi, vf);
        ctx[t] = __builtin_amdgcn_wmma_f32_16x16x32_bf16(false, pa.v, false, vf.v,
                                                         (short)0, ctx[t], false, false);
      }
    }

    if (c + 1 < nch) wait_async0();           // next chunk resident
    __syncthreads();                          // all waves done with buf before overwrite
  }

  #pragma unroll
  for (int t = 0; t < 8; t++)
    #pragma unroll
    for (int r = 0; r < 8; r++) {
      size_t idx = ((size_t)b * TSEQ + t0 + r + 8 * hi) * (size_t)DIM + h * DHD + t * 16 + ln;
      ctxb[idx] = f2bf(ctx[t][r] / lrow[r]);
    }
}

// ---------------- launch ----------------
extern "C" void kernel_launch(void* const* d_in, const int* in_sizes, int n_in,
                              void* d_out, int out_size, void* d_ws, size_t ws_size,
                              hipStream_t stream) {
  const float* x    = (const float*)d_in[0];
  const float* Wqd  = (const float*)d_in[1];
  const float* Wkvd = (const float*)d_in[2];
  const float* Wqu  = (const float*)d_in[3];
  const float* Wku  = (const float*)d_in[4];
  const float* Wvu  = (const float*)d_in[5];
  const float* Wo   = (const float*)d_in[6];
  const float* bo   = (const float*)d_in[7];
  float* out = (float*)d_out;
  (void)in_sizes; (void)n_in; (void)out_size; (void)ws_size;

  char* ws = (char*)d_ws;
  size_t off = 0;
  auto alloc = [&](size_t elems) {
    u16* p = (u16*)(ws + off);
    off += (elems * sizeof(u16) + 255) & ~(size_t)255;
    return p;
  };
  u16* xb     = alloc((size_t)BBATCH * TSEQ * DIM);       // 16 MB
  u16* wqd_b  = alloc((size_t)RQD * DIM);                 //  4 MB
  u16* wkvd_b = alloc((size_t)RKD * DIM);                 //  2 MB
  u16* uqT    = alloc((size_t)NH * RQD * DHD);            //  4 MB
  u16* ukT    = alloc((size_t)NH * RKD * DHD);            //  2 MB
  u16* wv_b   = alloc((size_t)DIM * RKD);                 //  2 MB
  u16* wo_b   = alloc((size_t)DIM * DIM);                 //  8 MB
  u16* qlat   = alloc((size_t)BBATCH * TSEQ * RQD);       //  8 MB
  u16* ckv    = alloc((size_t)BBATCH * TSEQ * RKD);       //  4 MB
  u16* mmT    = alloc((size_t)NH * RKD * RQD);            // 16 MB
  u16* qm     = alloc((size_t)BBATCH * NH * TSEQ * RKD);  // 64 MB
  u16* vupT   = alloc((size_t)BBATCH * NH * DHD * TSEQ);  // 16 MB
  u16* ctxb   = alloc((size_t)BBATCH * TSEQ * DIM);       // 16 MB

  auto cvt = [&](const float* s, u16* d, long n) {
    int blocks = (int)((n + 2047) / 2048);
    k_f32_to_bf16<<<dim3(blocks), dim3(256), 0, stream>>>(s, d, n);
  };
  cvt(x,    xb,     (long)BBATCH * TSEQ * DIM);
  cvt(Wqd,  wqd_b,  (long)RQD * DIM);
  cvt(Wkvd, wkvd_b, (long)RKD * DIM);
  cvt(Wvu,  wv_b,   (long)DIM * RKD);
  cvt(Wo,   wo_b,   (long)DIM * DIM);
  {
    long n = (long)NH * RQD * DHD;
    k_head_transpose<<<dim3((int)((n + 2047) / 2048)), dim3(256), 0, stream>>>(Wqu, uqT, RQD, n);
    n = (long)NH * RKD * DHD;
    k_head_transpose<<<dim3((int)((n + 2047) / 2048)), dim3(256), 0, stream>>>(Wku, ukT, RKD, n);
  }

  const int BIG = 1 << 30;
  // 1) Q_lat = xb * Wqd^T            [4096,1024]
  k_gemm_nt<<<dim3(RQD / 64, (BBATCH * TSEQ) / 128, 1), dim3(256), 0, stream>>>(
      xb, wqd_b, qlat, DIM, DIM, DIM, RQD, 0, 1, 1, 0, 1, 1, 0, 1.0f, nullptr, 0);
  // 2) C_kv = xb * Wkvd^T            [4096,512]
  k_gemm_nt<<<dim3(RKD / 64, (BBATCH * TSEQ) / 128, 1), dim3(256), 0, stream>>>(
      xb, wkvd_b, ckv, DIM, DIM, DIM, RKD, 0, 1, 1, 0, 1, 1, 0, 1.0f, nullptr, 0);
  // 3) M^T[h] = UkT[h] * UqT[h]^T / sqrt(DH)   [512,1024] x16
  k_gemm_nt<<<dim3(RQD / 64, RKD / 128, NH), dim3(256), 0, stream>>>(
      ukT, uqT, mmT, DHD, DHD, DHD, RQD,
      (long)RKD * DHD, 1, NH, (long)RQD * DHD, 1, NH, (long)RKD * RQD,
      0.088388347648318447f, nullptr, 0);
  // 4) Qm[b,h] = Qlat[b] * M^T[h]^T  [2048,512] x32
  k_gemm_nt<<<dim3(RKD / 64, TSEQ / 128, BBATCH * NH), dim3(256), 0, stream>>>(
      qlat, mmT, qm, RQD, RQD, RQD, RKD,
      (long)TSEQ * RQD, NH, BIG, (long)RKD * RQD, 1, NH, (long)TSEQ * RKD,
      1.0f, nullptr, 0);
  // 5) Vup^T[b,h] = Wv_h * Ckv[b]^T  [128,2048] x32  (transposed for flash B-frags)
  k_gemm_nt<<<dim3(TSEQ / 64, DHD / 128, BBATCH * NH), dim3(256), 0, stream>>>(
      wv_b, ckv, vupT, RKD, RKD, RKD, TSEQ,
      (long)DHD * RKD, 1, NH, (long)TSEQ * RKD, NH, BIG, (long)DHD * TSEQ,
      1.0f, nullptr, 0);
  // 6) flash attention -> ctx (bf16, [B,T,H*DH]); 88KB dynamic LDS, async-staged K/V
  k_flash<<<dim3(TSEQ / 128, BBATCH * NH), dim3(256), FLASH_LDS_BYTES, stream>>>(
      qm, ckv, vupT, ctxb);
  // 7) out = ctx * Wo^T + bo  (fp32)
  k_gemm_nt<<<dim3(DIM / 64, (BBATCH * TSEQ) / 128, 1), dim3(256), 0, stream>>>(
      ctxb, wo_b, out, DIM, DIM, DIM, DIM, 0, 1, 1, 0, 1, 1, 0, 1.0f, bo, 1);
}
